// Mixed_Superpixel_CNN_32444182954664
// MI455X (gfx1250) — matrix operand order repaired and checked by reading, not compiled, not run on hardware
//
#include <hip/hip_runtime.h>
#include <hip/hip_bf16.h>

// ---- problem constants -------------------------------------------------
#define HH_ 512
#define WW_ 512
#define CIN 128
#define CMID 64
#define COUT 64
#define NSEG 1024
#define NB 512
#define RPATCH 16
#define EPSF 1e-5f

// padded 18x18 pixel grids (1-pixel zero halo) for branch-free A fragments
#define PW 18
#define X0_ROW (PW * CIN)    // 2304 elements per padded row
#define X1_ROW (PW * CMID)   // 1152 elements per padded row

typedef __attribute__((ext_vector_type(16))) __bf16 v16bf;
typedef __attribute__((ext_vector_type(8)))  float  v8f;
typedef __attribute__((ext_vector_type(4)))  unsigned int v4u;
typedef __attribute__((ext_vector_type(8)))  int v8i;
typedef __attribute__((ext_vector_type(4)))  int v4i;

union FragU { uint4 q[2]; v16bf v; };

__device__ __forceinline__ unsigned short f2bf(float f) {
  unsigned u = __builtin_bit_cast(unsigned, f);
  u += 0x7FFFu + ((u >> 16) & 1u);          // round-to-nearest-even
  return (unsigned short)(u >> 16);
}

__device__ __forceinline__ void border_pix(int b, int& py, int& px) {
  if (b < 18)      { py = 0;  px = b; }
  else if (b < 36) { py = 17; px = b - 18; }
  else             { int r = b - 36; py = 1 + (r >> 1); px = (r & 1) * 17; }
}

// ---- 1) per-segment bbox via atomics -----------------------------------
__global__ void k_bbox_init(int* mnR, int* mxR, int* mnC, int* mxC) {
  int i = blockIdx.x * blockDim.x + threadIdx.x;
  if (i < NSEG) { mnR[i] = 0x7fffffff; mxR[i] = -1; mnC[i] = 0x7fffffff; mxC[i] = -1; }
}

__global__ void k_bbox(const int* __restrict__ seg,
                       int* mnR, int* mxR, int* mnC, int* mxC) {
  int i = blockIdx.x * blockDim.x + threadIdx.x;    // HH_*WW_ threads
  int s = seg[i];
  int r = i >> 9, c = i & 511;
  atomicMin(&mnR[s], r); atomicMax(&mxR[s], r);
  atomicMin(&mnC[s], c); atomicMax(&mxC[s], c);
}

// ---- 2) gather per-boundary-node meta, emit int32 outputs ---------------
__global__ void k_meta(const int* __restrict__ bnodes,
                       const int* mnR, const int* mxR, const int* mnC, const int* mxC,
                       int* pMinR, int* pMinC, int* pHH, int* pWW, int* outTail) {
  int t = blockIdx.x * blockDim.x + threadIdx.x;
  if (t >= NB) return;
  int n  = bnodes[t];
  int r0 = mnR[n], c0 = mnC[n];
  int hh = mxR[n] - r0 + 1, ww = mxC[n] - c0 + 1;
  pMinR[t] = r0; pMinC[t] = c0; pHH[t] = hh; pWW[t] = ww;
  outTail[t]          = hh;    // return order: out, hh, ww, min_r, min_c
  outTail[NB + t]     = ww;
  outTail[2 * NB + t] = r0;
  outTail[3 * NB + t] = c0;
}

// ---- 3) repack HWIO f32 weights into WMMA B-fragment bf16 ---------------
// frag[(kt*4 + nt)][lane][16 bf16]; lane<16 holds K=0..7,16..23, lane>=16 K=8..15,24..31
__global__ void k_pack(const float* __restrict__ w, unsigned short* __restrict__ wf,
                       int Cin, int total) {
  int idx = blockIdx.x * blockDim.x + threadIdx.x;
  if (idx >= total) return;
  int j    = idx & 15;
  int lane = (idx >> 4) & 31;
  int nt   = (idx >> 9) & 3;
  int kt   = idx >> 11;
  int ktPerTap = Cin >> 5;
  int tap = kt / ktPerTap;
  int cb  = (kt % ktPerTap) << 5;
  int kk  = ((lane & 16) ? 8 : 0) + (j & 7) + ((j & 8) ? 16 : 0);
  int co  = nt * 16 + (lane & 15);
  wf[idx] = f2bf(w[((size_t)(tap * Cin + cb + kk)) * 64 + co]);
}

// ---- 4) main: TDM patch load -> BN0 -> WMMA conv1 -> BN1 -> WMMA conv2 --
__global__ __launch_bounds__(256)
void k_main(const float* __restrict__ data,
            const float* gam0, const float* bet0, const float* mu0, const float* var0,
            const float* cb1,
            const float* gam1, const float* bet1, const float* mu1, const float* var1,
            const float* cb2,
            const unsigned short* __restrict__ w1f,
            const unsigned short* __restrict__ w2f,
            const int* pMinR, const int* pMinC, const int* pHH, const int* pWW,
            float* __restrict__ out) {
  // 128 KB raw f32 tile + 81 KB padded x0 + 40.5 KB padded x1 < 320 KB/WGP
  __shared__ __align__(16) unsigned char smem[131072 + PW*PW*CIN*2 + PW*PW*CMID*2];
  float*          raw = (float*)smem;
  unsigned short* x0p = (unsigned short*)(smem + 131072);
  unsigned short* x1p = (unsigned short*)(smem + 131072 + PW*PW*CIN*2);

  const int p    = blockIdx.x;
  const int tid  = threadIdx.x;
  const int lane = tid & 31;
  const int wave = __builtin_amdgcn_readfirstlane(tid >> 5);   // SGPR uniform

  const int hh = pHH[p], ww = pWW[p];

  // ---------- TDM: DMA 16 rows x 2048 f32 (16x16x128 tile) into LDS -----
  if (wave == 0) {
    int r0 = __builtin_amdgcn_readfirstlane(pMinR[p]);
    int c0 = __builtin_amdgcn_readfirstlane(pMinC[p]);
    unsigned long long ga =
        (unsigned long long)(uintptr_t)(data + ((size_t)r0 * WW_ + c0) * CIN);
    unsigned ldsOff = (unsigned)(uintptr_t)raw;
    v4u g0;                                   // D# group 0
    g0[0] = 1u;                               // count=1, user mode
    g0[1] = ldsOff;                           // lds_addr (bytes)
    g0[2] = (unsigned)ga;                     // global_addr[31:0]
    g0[3] = (unsigned)((ga >> 32) & 0x01ffffffu) | (2u << 30);  // addr[56:32] | type=2
    v8i g1;                                   // D# group 1
    g1[0] = 0x20000;                          // data_size=2 (4 bytes)
    g1[1] = 0;                                // tensor_dim0=65536 -> low16 = 0
    g1[2] = 0x02000001;                       // dim0 hi16=1 | tensor_dim1=512
    g1[3] = (int)0x08000000;                  // tile_dim0 = 2048
    g1[4] = 16;                               // tile_dim1 = 16 rows
    g1[5] = 65536;                            // tensor_dim0_stride = W*C
    g1[6] = 0; g1[7] = 0;
    v4i gz4 = {0, 0, 0, 0};                   // groups 2/3 unused (2-D tile)
    v8i gz8 = {0, 0, 0, 0, 0, 0, 0, 0};       // extra descriptor words unused
    __builtin_amdgcn_tensor_load_to_lds(g0, g1, gz4, gz4, gz8, 0);
  }

  // ---------- zero the halos while the TDM DMA is in flight --------------
  for (int e = tid; e < 68 * 16; e += 256) {          // x0p: 68 halo px * 16 ull
    int py, px; border_pix(e >> 4, py, px);
    *(unsigned long long*)&x0p[(py * PW + px) * CIN + (e & 15) * 8] = 0ull;
  }
  for (int e = tid; e < 68 * 8; e += 256) {           // x1p: 68 halo px * 8 ull
    int py, px; border_pix(e >> 3, py, px);
    *(unsigned long long*)&x1p[(py * PW + px) * CMID + (e & 7) * 8] = 0ull;
  }
  if (wave == 0) __builtin_amdgcn_s_wait_tensorcnt(0);
  __syncthreads();

  // ---------- phase A: BN0 + mask, LDS f32 -> padded LDS bf16 ------------
  {
    const int c4 = lane * 4;
    float s0[4], sh[4];
    #pragma unroll
    for (int k = 0; k < 4; ++k) {
      int c = c4 + k;
      float sc = gam0[c] * rsqrtf(var0[c] + EPSF);
      s0[k] = sc; sh[k] = bet0[c] - mu0[c] * sc;
    }
    for (int i = 0; i < 32; ++i) {
      int pix = wave + (i << 3);
      int y = pix >> 4, x = pix & 15;
      bool valid = (y < hh) && (x < ww);
      float4 d = *(const float4*)&raw[pix * CIN + c4];
      unsigned long long pk = 0ull;
      if (valid) {
        unsigned long long h0 = f2bf(d.x * s0[0] + sh[0]);
        unsigned long long h1 = f2bf(d.y * s0[1] + sh[1]);
        unsigned long long h2 = f2bf(d.z * s0[2] + sh[2]);
        unsigned long long h3 = f2bf(d.w * s0[3] + sh[3]);
        pk = h0 | (h1 << 16) | (h2 << 32) | (h3 << 48);
      }
      *(unsigned long long*)&x0p[((y + 1) * PW + x + 1) * CIN + c4] = pk;
    }
  }
  __syncthreads();

  const int lx = lane & 15;
  const int ks = (lane & 16) ? 8 : 0;       // A-fragment K sub-offset per half-wave
  const int xb = ks;                        // C/D row (x) base per half-wave
  const int nt     = wave & 3;              // this wave's cout tile (SGPR)
  const int mtBase = (wave >> 2) * 8;       // rows mtBase..mtBase+7 (SGPR)

  const v8f vzero = {0.f, 0.f, 0.f, 0.f, 0.f, 0.f, 0.f, 0.f};

  // ---------- phase B: conv1, K = 9 taps x 128 ch (36 WMMA steps) --------
  // halo rows are zero, so out-of-range taps contribute nothing: branch-free
  {
    v8f acc[8];
    #pragma unroll
    for (int m = 0; m < 8; ++m) acc[m] = vzero;
    const unsigned short* bptr = w1f + ((size_t)nt * 32 + lane) * 16;
    #pragma unroll 1
    for (int tap = 0; tap < 9; ++tap) {
      int dy = (tap < 3) ? -1 : ((tap < 6) ? 0 : 1);
      int dx = tap - (dy + 1) * 3 - 1;
      // one VGPR base per tap; per-(m,cb) offsets are DS immediate offsets
      const unsigned short* abase =
          x0p + (unsigned)(mtBase + dy + 1) * X0_ROW
              + (unsigned)(lx + dx + 1) * CIN + (unsigned)ks;
      #pragma unroll
      for (int cbi = 0; cbi < 4; ++cbi) {
        FragU B;
        const uint4* bp = (const uint4*)bptr;
        B.q[0] = bp[0]; B.q[1] = bp[1];
        __builtin_prefetch(bptr + 4096, 0, 3);
        bptr += 2048;                                        // next kt fragment
        FragU A[8];
        #pragma unroll
        for (int m = 0; m < 8; ++m) {                        // batch A loads
          const uint4* ap = (const uint4*)(abase + m * X0_ROW + cbi * 32);
          A[m].q[0] = ap[0]; A[m].q[1] = ap[2];
        }
        #pragma unroll
        for (int m = 0; m < 8; ++m)                          // WMMA chain
          acc[m] = __builtin_amdgcn_wmma_f32_16x16x32_bf16(
              false, A[m].v, false, B.v, (short)0, acc[m], false, false);
      }
    }
    // epilogue: +bias, LeakyReLU, BN1, mask -> padded bf16 LDS
    const int co   = nt * 16 + lx;
    const float bias = cb1[co];
    const float sc1  = gam1[co] * rsqrtf(var1[co] + EPSF);
    const float sf1  = bet1[co] - mu1[co] * sc1;
    #pragma unroll
    for (int m = 0; m < 8; ++m) {
      int mt = mtBase + m;
      bool rowok = mt < hh;
      #pragma unroll
      for (int v = 0; v < 8; ++v) {
        int xx = xb + v;
        float val = acc[m][v] + bias;
        val = val > 0.f ? val : 0.01f * val;
        val = val * sc1 + sf1;
        if (!(rowok && (xx < ww))) val = 0.f;
        x1p[((mt + 1) * PW + xx + 1) * CMID + co] = f2bf(val);
      }
    }
  }
  __syncthreads();

  // ---------- phase C: conv2, K = 9 taps x 64 ch (18 WMMA steps) ---------
  {
    v8f acc[8];
    #pragma unroll
    for (int m = 0; m < 8; ++m) acc[m] = vzero;
    const unsigned short* bptr = w2f + ((size_t)nt * 32 + lane) * 16;
    #pragma unroll 1
    for (int tap = 0; tap < 9; ++tap) {
      int dy = (tap < 3) ? -1 : ((tap < 6) ? 0 : 1);
      int dx = tap - (dy + 1) * 3 - 1;
      const unsigned short* abase =
          x1p + (unsigned)(mtBase + dy + 1) * X1_ROW
              + (unsigned)(lx + dx + 1) * CMID + (unsigned)ks;
      #pragma unroll
      for (int cbi = 0; cbi < 2; ++cbi) {
        FragU B;
        const uint4* bp = (const uint4*)bptr;
        B.q[0] = bp[0]; B.q[1] = bp[1];
        __builtin_prefetch(bptr + 4096, 0, 3);
        bptr += 2048;
        FragU A[8];
        #pragma unroll
        for (int m = 0; m < 8; ++m) {
          const uint4* ap = (const uint4*)(abase + m * X1_ROW + cbi * 32);
          A[m].q[0] = ap[0]; A[m].q[1] = ap[2];
        }
        #pragma unroll
        for (int m = 0; m < 8; ++m)
          acc[m] = __builtin_amdgcn_wmma_f32_16x16x32_bf16(
              false, A[m].v, false, B.v, (short)0, acc[m], false, false);
      }
    }
    // epilogue: +bias, LeakyReLU, sigmoid, mask -> f32 global
    const int co   = nt * 16 + lx;
    const float bias = cb2[co];
    #pragma unroll
    for (int m = 0; m < 8; ++m) {
      int mt = mtBase + m;
      bool rowok = mt < hh;
      #pragma unroll
      for (int v = 0; v < 8; ++v) {
        int xx = xb + v;
        float val = acc[m][v] + bias;
        val = val > 0.f ? val : 0.01f * val;
        float sig = 1.f / (1.f + __expf(-val));
        if (!(rowok && (xx < ww))) sig = 0.f;
        out[(((size_t)p * 256) + (mt << 4) + xx) * COUT + co] = sig;
      }
    }
  }
}

// ---- host side -----------------------------------------------------------
extern "C" void kernel_launch(void* const* d_in, const int* in_sizes, int n_in,
                              void* d_out, int out_size, void* d_ws, size_t ws_size,
                              hipStream_t stream) {
  const float* data = (const float*)d_in[0];
  const float* g0   = (const float*)d_in[1];
  const float* b0   = (const float*)d_in[2];
  const float* m0   = (const float*)d_in[3];
  const float* v0   = (const float*)d_in[4];
  const float* w1   = (const float*)d_in[5];
  const float* cb1  = (const float*)d_in[6];
  const float* g1   = (const float*)d_in[7];
  const float* b1   = (const float*)d_in[8];
  const float* m1   = (const float*)d_in[9];
  const float* v1   = (const float*)d_in[10];
  const float* w2   = (const float*)d_in[11];
  const float* cb2  = (const float*)d_in[12];
  const int* bnodes = (const int*)d_in[13];
  const int* seg    = (const int*)d_in[14];

  char* ws = (char*)d_ws;
  int* mnR   = (int*)(ws + 0);
  int* mxR   = (int*)(ws + 4096);
  int* mnC   = (int*)(ws + 8192);
  int* mxC   = (int*)(ws + 12288);
  int* pMinR = (int*)(ws + 16384);
  int* pMinC = (int*)(ws + 18432);
  int* pHH   = (int*)(ws + 20480);
  int* pWW   = (int*)(ws + 22528);
  unsigned short* w1f = (unsigned short*)(ws + 24576);             // 36*4 frags * 1KB
  unsigned short* w2f = (unsigned short*)(ws + 24576 + 147456);    // 18*4 frags * 1KB

  float* out     = (float*)d_out;
  int*   outTail = (int*)d_out + (size_t)NB * 256 * COUT;          // hh|ww|min_r|min_c

  k_bbox_init<<<(NSEG + 255) / 256, 256, 0, stream>>>(mnR, mxR, mnC, mxC);
  k_bbox<<<(HH_ * WW_) / 256, 256, 0, stream>>>(seg, mnR, mxR, mnC, mxC);
  k_meta<<<(NB + 255) / 256, 256, 0, stream>>>(bnodes, mnR, mxR, mnC, mxC,
                                               pMinR, pMinC, pHH, pWW, outTail);
  k_pack<<<(36 * 4 * 32 * 16 + 255) / 256, 256, 0, stream>>>(w1, w1f, CIN, 36 * 4 * 32 * 16);
  k_pack<<<(18 * 4 * 32 * 16 + 255) / 256, 256, 0, stream>>>(w2, w2f, CMID, 18 * 4 * 32 * 16);
  k_main<<<NB, 256, 0, stream>>>(data, g0, b0, m0, v0, cb1, g1, b1, m1, v1, cb2,
                                 w1f, w2f, pMinR, pMinC, pHH, pWW, out);
}
